// Decoder_89163521065801
// MI455X (gfx1250) — compile-verified, matrix operand order checked by reading, not run on hardware
//
#include <hip/hip_runtime.h>
#include <hip/hip_bf16.h>
#include <math.h>

// Problem dims (fixed by reference)
#define B_  32
#define T_  128
#define V_  32000
#define E_  512
#define H_  1024
#define SOS 1

typedef __bf16 bf16;
typedef __bf16 v16bf __attribute__((ext_vector_type(16)));
typedef float  v8f   __attribute__((ext_vector_type(8)));
typedef float  v4f   __attribute__((ext_vector_type(4)));
typedef unsigned int v4u __attribute__((ext_vector_type(4)));
typedef int v4i __attribute__((ext_vector_type(4)));

union Frag { v16bf v; v4u u[2]; };
union Acc  { v8f v; float f[8]; };

__device__ __forceinline__ v8f vzero() { v8f z = {0,0,0,0,0,0,0,0}; return z; }

__device__ __forceinline__ unsigned short f2bf_bits(float f) {
  unsigned u = __float_as_uint(f);
  unsigned r = u + 0x7FFFu + ((u >> 16) & 1u);   // round-to-nearest-even
  return (unsigned short)(r >> 16);
}
__device__ __forceinline__ bf16 f2bf(float f) {
  unsigned short s = f2bf_bits(f);
  return __builtin_bit_cast(bf16, s);
}

__device__ __forceinline__ v8f wmma_bf16(v16bf a, v16bf b, v8f c) {
  // D = A(16x32 bf16) * B(32x16 bf16) + C(16x16 f32)
  return __builtin_amdgcn_wmma_f32_16x16x32_bf16(false, a, false, b, (short)0, c,
                                                 false, false);
}

// ---- CDNA5 async global->LDS staging (ASYNCcnt path), guarded fallback ----
#if __has_builtin(__builtin_amdgcn_global_load_async_to_lds_b128)
#define USE_ASYNC_LDS 1
typedef __attribute__((address_space(3))) v4i as3_v4i;
__device__ __forceinline__ void async_cp16(const void* g, void* l) {
  __builtin_amdgcn_global_load_async_to_lds_b128((v4i*)g, (as3_v4i*)l, 0, 0);
}
__device__ __forceinline__ void async_wait0() {
#if __has_builtin(__builtin_amdgcn_s_wait_asynccnt)
  __builtin_amdgcn_s_wait_asynccnt(0);
#else
  asm volatile("s_wait_asynccnt 0x0" ::: "memory");
#endif
}
#else
#define USE_ASYNC_LDS 0
#endif

// ---------------------------------------------------------------------------
// fp32 -> bf16 conversion, 4 elements / thread / iteration
// ---------------------------------------------------------------------------
__global__ void k_cvt4(const float* __restrict__ src, bf16* __restrict__ dst, int n4) {
  int i = blockIdx.x * blockDim.x + threadIdx.x;
  int stride = gridDim.x * blockDim.x;
  for (; i < n4; i += stride) {
    v4f f = *(const v4f*)(src + (size_t)i * 4);
    unsigned long long w =
        (unsigned long long)f2bf_bits(f.x)
      | ((unsigned long long)f2bf_bits(f.y) << 16)
      | ((unsigned long long)f2bf_bits(f.z) << 32)
      | ((unsigned long long)f2bf_bits(f.w) << 48);
    *(unsigned long long*)(dst + (size_t)i * 4) = w;
  }
}

// ---------------------------------------------------------------------------
// Embedding gather: tokens = [SOS, target[:,1:]] ; X[b*T+t] = bf16(emb[tok])
// ---------------------------------------------------------------------------
__global__ void k_gather(const int* __restrict__ target, const float* __restrict__ emb,
                         bf16* __restrict__ X) {
  const int row = blockIdx.x;            // b*T + t
  const int t = row % T_;
  const int tok = (t == 0) ? SOS : target[row];
  const float* e = emb + (size_t)tok * E_;
  bf16* x = X + (size_t)row * E_;
  for (int i = threadIdx.x; i < E_ / 4; i += blockDim.x) {
    v4f f = *(const v4f*)(e + (size_t)i * 4);
    unsigned long long w =
        (unsigned long long)f2bf_bits(f.x)
      | ((unsigned long long)f2bf_bits(f.y) << 16)
      | ((unsigned long long)f2bf_bits(f.z) << 32)
      | ((unsigned long long)f2bf_bits(f.w) << 48);
    *(unsigned long long*)(x + (size_t)i * 4) = w;
  }
}

// ---------------------------------------------------------------------------
// Input projection (batched over all timesteps, removed from serial path):
// Xproj(4096,1024) = X(4096,512) @ W_ih^T + b_ih + b_hh   (fp32 accum/out)
// One 16x16 output tile per wave, 8 waves / block.
// ---------------------------------------------------------------------------
__global__ void __launch_bounds__(256)
k_inproj(const bf16* __restrict__ X, const bf16* __restrict__ Wih,
         const float* __restrict__ b_ih, const float* __restrict__ b_hh,
         float* __restrict__ Xproj) {
  const int wave = threadIdx.x >> 5;
  const int lane = threadIdx.x & 31;
  const int tile = blockIdx.x * 8 + wave;     // 256*64 = 16384 tiles
  const int mtile = tile >> 6;                // M tiles: 4096/16
  const int ntile = tile & 63;                // N tiles: 1024/16
  const int m0 = mtile * 16, n0 = ntile * 16;
  const int lr = lane & 15;
  const int kb = (lane >> 4) * 8;             // K half-select per ISA A/B layout
  const bf16* Arow = X   + (size_t)(m0 + lr) * E_ + kb;
  const bf16* Brow = Wih + (size_t)(n0 + lr) * E_ + kb;   // B col n = W_ih row n
  Acc c; c.v = vzero();
  for (int k0 = 0; k0 < E_; k0 += 32) {
    Frag a, b;
    a.u[0] = *(const v4u*)(Arow + k0);
    a.u[1] = *(const v4u*)(Arow + k0 + 16);
    b.u[0] = *(const v4u*)(Brow + k0);
    b.u[1] = *(const v4u*)(Brow + k0 + 16);
    c.v = wmma_bf16(a.v, b.v, c.v);
  }
  const int col = n0 + lr;
  const float bias = b_ih[col] + b_hh[col];
  const int rbase = m0 + ((lane >> 4) * 8);
  for (int j = 0; j < 8; ++j)
    Xproj[(size_t)(rbase + j) * H_ + col] = c.f[j] + bias;
}

// ---------------------------------------------------------------------------
// Recurrence: h_t = tanh(Xproj[:,t] + h_{t-1} @ W_hh^T)
// 64 persistent blocks; block g owns output columns [16g,16g+16).
// Its W_hh slice (16x1024 bf16) lives in LDS (padded stride to spread banks).
// h ping-pongs through L2 (32x1024 bf16 = 64KB); one global barrier per step.
// ---------------------------------------------------------------------------
#define RNN_BLOCKS 64
#define WSL_STRIDE_B 2064   // 1024 halves (2048B) + 16B pad -> conflict-free rows

__global__ void k_rnn(const bf16* __restrict__ Whh, const float* __restrict__ Xproj,
                      bf16* __restrict__ hA, bf16* __restrict__ hB,
                      bf16* __restrict__ Hall, unsigned* __restrict__ bar) {
  __shared__ __align__(16) char Wsl[16 * WSL_STRIDE_B];
  const int tid = threadIdx.x;                // 64 threads = 2 waves
  const int n0 = blockIdx.x * 16;
  // Stage this block's W_hh slice once (rows n0..n0+15, full K).
  for (int i = tid; i < 16 * 128; i += blockDim.x) {
    int r = i >> 7, q = i & 127;
    *(v4u*)(Wsl + r * WSL_STRIDE_B + q * 16) =
        *(const v4u*)(Whh + (size_t)(n0 + r) * H_ + q * 8);
  }
  __syncthreads();

  const int lane = tid & 31;
  const int mt = tid >> 5;                    // wave -> M tile (B rows 0..15 / 16..31)
  const int m0 = mt * 16;
  const int lr = lane & 15;
  const int kb = (lane >> 4) * 8;
  const int col = n0 + lr;
  const int rbase = m0 + ((lane >> 4) * 8);

  for (int t = 0; t < T_; ++t) {
    const bf16* hin = (t & 1) ? hB : hA;
    bf16* hout      = (t & 1) ? hA : hB;
    const bf16* Arow = hin + (size_t)(m0 + lr) * H_ + kb;
    Acc c; c.v = vzero();
    for (int k0 = 0; k0 < H_; k0 += 32) {
      Frag a, b;
      a.u[0] = *(const v4u*)(Arow + k0);
      a.u[1] = *(const v4u*)(Arow + k0 + 16);
      const char* bp = Wsl + lr * WSL_STRIDE_B + (size_t)(k0 + kb) * 2;
      b.u[0] = *(const v4u*)bp;
      b.u[1] = *(const v4u*)(bp + 32);
      c.v = wmma_bf16(a.v, b.v, c.v);
    }
    for (int j = 0; j < 8; ++j) {
      const int bi = rbase + j;               // batch row
      float v = tanhf(c.f[j] + Xproj[(size_t)(bi * T_ + t) * H_ + col]);
      bf16 hv = f2bf(v);
      hout[(size_t)bi * H_ + col] = hv;
      Hall[(size_t)(bi * T_ + t) * H_ + col] = hv;   // history for decode GEMM
    }
    // ---- sense-reversal grid barrier over 64 blocks ----
    __syncthreads();
    if (tid == 0) {
      unsigned g = __hip_atomic_load(&bar[1], __ATOMIC_ACQUIRE, __HIP_MEMORY_SCOPE_AGENT);
      unsigned v = __hip_atomic_fetch_add(&bar[0], 1u, __ATOMIC_ACQ_REL, __HIP_MEMORY_SCOPE_AGENT);
      if (v == RNN_BLOCKS - 1u) {
        __hip_atomic_store(&bar[0], 0u, __ATOMIC_RELAXED, __HIP_MEMORY_SCOPE_AGENT);
        __hip_atomic_fetch_add(&bar[1], 1u, __ATOMIC_RELEASE, __HIP_MEMORY_SCOPE_AGENT);
      } else {
        while (__hip_atomic_load(&bar[1], __ATOMIC_ACQUIRE, __HIP_MEMORY_SCOPE_AGENT) == g)
          __builtin_amdgcn_s_sleep(2);
      }
    }
    __syncthreads();
  }
}

// ---------------------------------------------------------------------------
// Decode GEMM (dominant: 268 GFLOP):
// out(4096,32000) = Hall(4096,1024 bf16) @ W_out^T + b_out   (fp32 out)
// Block tile 64x256, 8 waves in 2x4 grid, each wave 32x64 (8 accum tiles).
// K staged in 32-wide chunks through double-buffered LDS (one barrier/chunk);
// staging uses GLOBAL_LOAD_ASYNC_TO_LDS when available. Output stores are
// non-temporal (write-once 524MB stream; keep W_out/Hall resident in L2).
// ---------------------------------------------------------------------------
#define BM 64
#define BN 256
#define KC 32
#define LSTR 80                     // LDS row stride bytes: 64B data + 16B pad
#define BUF_BYTES ((BM + BN) * LSTR)

__global__ void __launch_bounds__(256)
k_outproj(const bf16* __restrict__ Hall, const bf16* __restrict__ Wout,
          const float* __restrict__ b_out, float* __restrict__ out) {
  __shared__ __align__(16) char lds[2 * BUF_BYTES];   // 2 x 25.6 KB
  const int tid = threadIdx.x;
  const int nb = V_ / BN;                 // 125
  const int bx = blockIdx.x % nb;
  const int by = blockIdx.x / nb;         // 0..63
  const size_t abase = (size_t)by * BM * H_;
  const size_t bbase = (size_t)bx * BN * H_;
  const int wave = tid >> 5, lane = tid & 31;
  const int wm = wave & 1, wn = wave >> 1;
  const int lr = lane & 15;
  const int kb = (lane >> 4) * 8;

  const int ar = tid >> 2, aq = tid & 3;  // A staging: 1 x 16B chunk / thread

  // stage one 32-wide K chunk of A(64 rows) + B(256 rows) into buffer `buf`
  auto stage = [&](int buf, int k0) {
    char* As = lds + buf * BUF_BYTES;
    char* Bs = As + BM * LSTR;
    const bf16* aSrc = Hall + abase + (size_t)ar * H_ + k0 + aq * 8;
#if USE_ASYNC_LDS
    async_cp16(aSrc, As + ar * LSTR + aq * 16);
#else
    *(v4u*)(As + ar * LSTR + aq * 16) = *(const v4u*)aSrc;
#endif
    for (int it = 0; it < 4; ++it) {      // B staging: 4 chunks / thread
      int idx = tid + it * 256;
      int r = idx >> 2, q = idx & 3;
      const bf16* bSrc = Wout + bbase + (size_t)r * H_ + k0 + q * 8;
#if USE_ASYNC_LDS
      async_cp16(bSrc, Bs + r * LSTR + q * 16);
#else
      *(v4u*)(Bs + r * LSTR + q * 16) = *(const v4u*)bSrc;
#endif
    }
  };

  Acc c[2][4];
  for (int mt = 0; mt < 2; ++mt)
    for (int nt = 0; nt < 4; ++nt) c[mt][nt].v = vzero();

  stage(0, 0);                            // prologue
  const int NK = H_ / KC;                 // 32 chunks
  for (int kk = 0; kk < NK; ++kk) {
#if USE_ASYNC_LDS
    async_wait0();                        // current buffer's async writes landed
#endif
    __syncthreads();                      // visible to all waves; prev compute done
    if (kk + 1 < NK) {
      stage((kk + 1) & 1, (kk + 1) * KC); // overlap staging with compute below
      __builtin_prefetch(Hall + abase + (size_t)ar * H_ + (kk + 1) * KC, 0, 0);
      __builtin_prefetch(Wout + bbase + (size_t)tid * H_ + (kk + 1) * KC, 0, 0);
    }
    const char* As = lds + (kk & 1) * BUF_BYTES;
    const char* Bs = As + BM * LSTR;
    Frag a[2], b;
    for (int mt = 0; mt < 2; ++mt) {
      const char* ap = As + (wm * 32 + mt * 16 + lr) * LSTR + kb * 2;
      a[mt].u[0] = *(const v4u*)ap;
      a[mt].u[1] = *(const v4u*)(ap + 32);
    }
    for (int nt = 0; nt < 4; ++nt) {
      const char* bp = Bs + (wn * 64 + nt * 16 + lr) * LSTR + kb * 2;
      b.u[0] = *(const v4u*)bp;
      b.u[1] = *(const v4u*)(bp + 32);
      c[0][nt].v = wmma_bf16(a[0].v, b.v, c[0][nt].v);
      c[1][nt].v = wmma_bf16(a[1].v, b.v, c[1][nt].v);
    }
  }

  for (int mt = 0; mt < 2; ++mt)
    for (int nt = 0; nt < 4; ++nt) {
      const int gcol = bx * BN + wn * 64 + nt * 16 + lr;
      const float bias = b_out[gcol];
      const int rbase = by * BM + wm * 32 + mt * 16 + ((lane >> 4) * 8);
      for (int j = 0; j < 8; ++j)
        __builtin_nontemporal_store(c[mt][nt].f[j] + bias,
                                    out + (size_t)(rbase + j) * V_ + gcol);
    }
}

// ---------------------------------------------------------------------------
extern "C" void kernel_launch(void* const* d_in, const int* in_sizes, int n_in,
                              void* d_out, int out_size, void* d_ws, size_t ws_size,
                              hipStream_t stream) {
  (void)in_sizes; (void)n_in; (void)out_size; (void)ws_size;
  const int*   target = (const int*)  d_in[0];
  const float* h0     = (const float*)d_in[1];
  const float* emb    = (const float*)d_in[2];
  const float* W_ih   = (const float*)d_in[3];
  const float* b_ih   = (const float*)d_in[4];
  const float* W_hh   = (const float*)d_in[5];
  const float* b_hh   = (const float*)d_in[6];
  const float* W_out  = (const float*)d_in[7];
  const float* b_out  = (const float*)d_in[8];
  float* out = (float*)d_out;

  // Workspace carve-up (~98 MB total)
  char* ws = (char*)d_ws;
  size_t off = 0;
  auto carve = [&](size_t bytes) -> char* {
    char* p = ws + off;
    off += (bytes + 255) & ~(size_t)255;
    return p;
  };
  bf16*     Wih_bf  = (bf16*)carve((size_t)H_ * E_ * 2);
  bf16*     Whh_bf  = (bf16*)carve((size_t)H_ * H_ * 2);
  bf16*     Wout_bf = (bf16*)carve((size_t)V_ * H_ * 2);
  bf16*     Xbf     = (bf16*)carve((size_t)B_ * T_ * E_ * 2);
  float*    Xproj   = (float*)carve((size_t)B_ * T_ * H_ * 4);
  bf16*     Hall    = (bf16*)carve((size_t)B_ * T_ * H_ * 2);
  bf16*     hA      = (bf16*)carve((size_t)B_ * H_ * 2);
  bf16*     hB      = (bf16*)carve((size_t)B_ * H_ * 2);
  unsigned* bar     = (unsigned*)carve(256);

  (void)hipMemsetAsync(bar, 0, 256, stream);

  k_cvt4<<<512,  256, 0, stream>>>(W_ih,  Wih_bf,  (H_ * E_) / 4);
  k_cvt4<<<512,  256, 0, stream>>>(W_hh,  Whh_bf,  (H_ * H_) / 4);
  k_cvt4<<<2048, 256, 0, stream>>>(W_out, Wout_bf, (V_ * H_) / 4);
  k_cvt4<<<64,   256, 0, stream>>>(h0,    hA,      (B_ * H_) / 4);
  k_gather<<<B_ * T_, 128, 0, stream>>>(target, emb, Xbf);
  k_inproj<<<2048, 256, 0, stream>>>(Xbf, Wih_bf, b_ih, b_hh, Xproj);
  k_rnn<<<RNN_BLOCKS, 64, 0, stream>>>(Whh_bf, Xproj, hA, hB, Hall, bar);
  k_outproj<<<(4096 / BM) * (V_ / BN), 256, 0, stream>>>(Hall, Wout_bf, b_out, out);
}
